// LSTMCRFModel_44487271252636
// MI455X (gfx1250) — compile-verified, matrix operand order checked by reading, not run on hardware
//
#include <hip/hip_runtime.h>
#include <hip/hip_bf16.h>
#include <math.h>

// ---------------------------------------------------------------------------
// BiLSTM-CRF forward for MI455X (gfx1250), wave32 + WMMA f16 (f32 accum)
// + Tensor Data Mover staging of the recurrent hidden state into LDS.
//   dims: V=30000 K=9 E=300 H=300 B=64 T=512 ; padded Ep=Hp=320, 4H->1280
// Workspace requirement ~446 MB (pre-activations dominate).
// ---------------------------------------------------------------------------

typedef _Float16 h16;
typedef __attribute__((ext_vector_type(16))) _Float16 v16h;
typedef __attribute__((ext_vector_type(8)))  float    v8f;

typedef __attribute__((ext_vector_type(4))) unsigned uint32x4;
typedef __attribute__((ext_vector_type(8))) int      int32x8;
typedef __attribute__((ext_vector_type(4))) int      int32x4;

namespace cfg {
constexpr int Kt   = 9;
constexpr int E    = 300;
constexpr int H    = 300;
constexpr int Bsz  = 64;
constexpr int T    = 512;
constexpr int Ep   = 320;            // padded E (K-dim, 10 x 32)
constexpr int Hp   = 320;            // padded H
constexpr int G4   = 1280;           // padded 4H (4 gates x 320)
constexpr int ROWS = T * Bsz;        // 32768 rows of the pre-GEMM
constexpr int NBLK = 5;              // persistent LSTM blocks per direction
}

// ----- WMMA helpers --------------------------------------------------------

union Frag16 { v16h v; unsigned u[8]; };

// Build a 16x32 f16 fragment (A-layout; also used for B = W^T since W is
// row-major NxK).  ISA 7.12.2: lane L holds row L%16; VGPR v holds K pair
// { (v>=4?16:0) + (L>=16?8:0) + (v%4)*2 } — pairs are contiguous dword loads
// (the compiler coalesces the 8 dwords into two b128s).
__device__ inline v16h load_frag_k(const h16* __restrict__ base, int ld,
                                   int row, int kbase, int lane) {
  Frag16 f;
  const int khi = (lane & 16) ? 8 : 0;
  const h16* p = base + row * ld + kbase + khi;
#pragma unroll
  for (int v = 0; v < 8; ++v) {
    const int k = ((v >= 4) ? 16 : 0) + (v & 3) * 2;
    f.u[v] = *reinterpret_cast<const unsigned*>(p + k);
  }
  return f.v;
}

__device__ inline v8f wmma_f16(v16h a, v16h b, v8f c) {
  // (neg_a, A, neg_b, B, c_mod, C, reuse_a, reuse_b)
  return __builtin_amdgcn_wmma_f32_16x16x32_f16(false, a, false, b,
                                                (short)0, c, false, false);
}

__device__ inline float sigmoidf_(float x) { return 1.f / (1.f + __expf(-x)); }

__device__ inline void wait_tensorcnt0() {
#if __has_builtin(__builtin_amdgcn_s_wait_tensorcnt)
  __builtin_amdgcn_s_wait_tensorcnt(0);
#else
  asm volatile("s_wait_tensorcnt 0x0" ::: "memory");
#endif
}

// ----- init: zero recurrent state + barrier words --------------------------

__global__ void init_kernel(h16* h16cur, float* ccur, unsigned* bar) {
  using namespace cfg;
  const int n = 2 * Bsz * Hp;
  for (int i = blockIdx.x * blockDim.x + threadIdx.x; i < n;
       i += gridDim.x * blockDim.x) {
    h16cur[i] = (h16)0.f;
    ccur[i] = 0.f;
    if (i < 8) bar[i] = 0u;
  }
}

// ----- weights: fp32 -> padded f16 (gate-chunk padded 300->320) ------------

__global__ void prep_weights_kernel(const float* Wihf, const float* Whhf,
                                    const float* bihf, const float* bhhf,
                                    const float* Wihb, const float* Whhb,
                                    const float* bihb, const float* bhhb,
                                    h16* WihP, h16* WhhP, float* bias) {
  using namespace cfg;
  const int NW = 2 * G4 * Ep;             // per weight class (E==H==300)
  const int total = 2 * NW + 2 * G4;
  for (int idx = blockIdx.x * blockDim.x + threadIdx.x; idx < total;
       idx += gridDim.x * blockDim.x) {
    if (idx < NW) {                       // W_ih padded
      const int d = idx / (G4 * Ep), r = idx % (G4 * Ep);
      const int n = r / Ep, k = r % Ep, g = n / 320, jj = n % 320;
      const float* W = d ? Wihb : Wihf;
      float v = (jj < H && k < E) ? W[(g * H + jj) * E + k] : 0.f;
      WihP[idx] = (h16)v;
    } else if (idx < 2 * NW) {            // W_hh padded
      const int i2 = idx - NW;
      const int d = i2 / (G4 * Hp), r = i2 % (G4 * Hp);
      const int n = r / Hp, k = r % Hp, g = n / 320, jj = n % 320;
      const float* W = d ? Whhb : Whhf;
      float v = (jj < H && k < H) ? W[(g * H + jj) * H + k] : 0.f;
      WhhP[i2] = (h16)v;
    } else {                              // combined bias, padded
      const int i3 = idx - 2 * NW;
      const int d = i3 / G4, n = i3 % G4, g = n / 320, jj = n % 320;
      float v = 0.f;
      if (jj < H)
        v = d ? (bihb[g * H + jj] + bhhb[g * H + jj])
              : (bihf[g * H + jj] + bhhf[g * H + jj]);
      bias[i3] = v;
    }
  }
}

// ----- embedding gather -> f16, (T,B,Ep) row-major, zero padded ------------

__global__ void embed_gather_kernel(const int* x, const float* emb, h16* embh) {
  using namespace cfg;
  const long long n = (long long)ROWS * Ep;
  for (long long i = (long long)blockIdx.x * blockDim.x + threadIdx.x; i < n;
       i += (long long)gridDim.x * blockDim.x) {
    const int r = (int)(i / Ep), e = (int)(i % Ep);
    const int t = r / Bsz, b = r % Bsz;
    float v = 0.f;
    if (e < E) {
      const int id = x[b * T + t];
      v = emb[(size_t)id * E + e];
    }
    embh[i] = (h16)v;
  }
}

// ----- pre = embh (32768x320) x WihP^T (->1280) + bias, both directions ----
// One wave computes a 16x64 patch; 10 k-steps of v_wmma_f32_16x16x32_f16.

__global__ void __launch_bounds__(256) gemm_pre_kernel(
    const h16* __restrict__ embh, const h16* __restrict__ WihP,
    const float* __restrict__ bias, float* __restrict__ pre) {
  using namespace cfg;
  const int lane = threadIdx.x & 31;
  const int gw = blockIdx.x * 8 + (threadIdx.x >> 5);
  const int PER_DIR = (ROWS / 16) * (G4 / 64);          // 2048 * 20
  const int dir = gw / PER_DIR;
  const int r = gw % PER_DIR;
  const int mt = r / (G4 / 64), ng = r % (G4 / 64);
  const h16* W = WihP + (size_t)dir * G4 * Ep;
  const float* bs = bias + dir * G4;
  float* out = pre + (size_t)dir * ROWS * G4;

  const int m0 = mt * 16, n0 = ng * 64;
  v8f acc[4] = {};
  const int arow = m0 + (lane & 15);
  for (int ks = 0; ks < Ep; ks += 32) {
    const v16h a = load_frag_k(embh, Ep, arow, ks, lane);
#pragma unroll
    for (int nt = 0; nt < 4; ++nt) {
      const v16h b = load_frag_k(W, Ep, n0 + nt * 16 + (lane & 15), ks, lane);
      acc[nt] = wmma_f16(a, b, acc[nt]);
    }
  }
  const int rlo = (lane >> 4) * 8, col = lane & 15;
#pragma unroll
  for (int nt = 0; nt < 4; ++nt) {
    const int n = n0 + nt * 16 + col;
    const float bv = bs[n];
#pragma unroll
    for (int rr = 0; rr < 8; ++rr) {
      const int row = m0 + rr + rlo;
      out[(size_t)row * G4 + n] = acc[nt][rr] + bv;
    }
  }
}

// ----- device-scope sense barrier for the persistent recurrence ------------

__device__ inline void dir_barrier(unsigned* cnt, unsigned* phs, unsigned nblk,
                                   unsigned& myphase) {
  __threadfence();
  __syncthreads();
  if (threadIdx.x == 0) {
    const unsigned old = atomicAdd(cnt, 1u);
    if (old == nblk - 1) {
      *cnt = 0u;
      __threadfence();
      atomicAdd(phs, 1u);
    } else {
      while (__hip_atomic_load(phs, __ATOMIC_ACQUIRE,
                               __HIP_MEMORY_SCOPE_AGENT) <= myphase) {
        __builtin_amdgcn_s_sleep(1);
      }
    }
  }
  __syncthreads();
  ++myphase;
}

// ----- TDM: stage h16cur (64x320 f16 = 40KB, contiguous) into LDS ----------
// D# group0: count=1, lds_addr, global_addr(57b), type=2 ("image").
// D# group1: data_size=2B (code 1), tensor_dim0=tile_dim0=20480 elements,
//            tensor_dim1=tile_dim1=1 row, stride0=20480.  Groups 2/3 unused
//            (zero-filled); this toolchain's builtin takes 6 args
//            (g0, g1, g2, g3, extra, cpol).

__device__ inline void tdm_stage_h(const h16* gsrc, h16* lds_dst) {
#if __has_builtin(__builtin_amdgcn_tensor_load_to_lds)
  using namespace cfg;
  const unsigned elems = (unsigned)(Bsz * Hp);          // 20480
  const unsigned long long ga = (unsigned long long)(uintptr_t)gsrc;
  uint32x4 g0;
  g0.x = 1u;                                            // count=1, user mode
  g0.y = (unsigned)(uintptr_t)lds_dst;                  // LDS byte address
  g0.z = (unsigned)ga;                                  // global_addr[31:0]
  g0.w = (unsigned)((ga >> 32) & 0x1FFFFFFu) | (2u << 30);  // [56:32] | type=2
  int32x8 g1;
  g1[0] = (int)(1u << 16);          // workgroup_mask=0, data_size=1 (2 bytes)
  g1[1] = (int)(elems << 16);       // tensor_dim0[15:0] << 16 (abar addr = 0)
  g1[2] = (int)(1u << 16);          // tensor_dim0[31:16]=0 | tensor_dim1=1
  g1[3] = (int)(elems << 16);       // tensor_dim1 hi=0 | tile_dim0=20480
  g1[4] = 1;                        // tile_dim1=1, tile_dim2=0
  g1[5] = (int)elems;               // tensor_dim0_stride[31:0]
  g1[6] = 0;                        // stride0 hi | stride1 lo
  g1[7] = 0;
  const int32x4 z4 = {0, 0, 0, 0};
  const int32x8 z8 = {0, 0, 0, 0, 0, 0, 0, 0};
  __builtin_amdgcn_tensor_load_to_lds(g0, g1, z4, z4, z8, 0);
  wait_tensorcnt0();
#else
  using namespace cfg;
  for (int e = (int)(threadIdx.x & 31); e < Bsz * Hp; e += 32)
    lds_dst[e] = gsrc[e];
#endif
}

// ----- persistent LSTM: grid = 10 blocks (5 per direction) -----------------
// Block owns hidden slice [blk*64, blk*64+64) => 256 gate columns.
// Per step: h_prev staged into LDS by the TDM, 4Mx16N tiles x 10 k-steps of
// WMMA against W_hh (L2-resident), gates spilled to a private global buffer,
// elementwise cell update (writes f16 h for the next TDM stage), global sync.

__global__ void __launch_bounds__(256, 1) lstm_kernel(
    const h16* __restrict__ WhhP, const float* __restrict__ pre,
    float* __restrict__ hs, h16* __restrict__ h16cur, float* __restrict__ ccur,
    float* __restrict__ gbuf, unsigned* bar) {
  using namespace cfg;
  __shared__ h16 h_lds[Bsz * Hp];                      // 40 KB

  const int dir = blockIdx.x / NBLK;
  const int blk = blockIdx.x % NBLK;
  const h16* W = WhhP + (size_t)dir * G4 * Hp;
  const float* preD = pre + (size_t)dir * ROWS * G4;
  float* hsD = hs + (size_t)dir * ROWS * Hp;
  h16* hc = h16cur + dir * Bsz * Hp;
  float* cc = ccur + dir * Bsz * Hp;
  float* gb = gbuf + (size_t)blockIdx.x * Bsz * 256;
  unsigned* cnt = bar + dir * 2;
  unsigned* phs = bar + dir * 2 + 1;

  const int tid = threadIdx.x, lane = tid & 31, wv = tid >> 5;
  const int mtile = wv & 3, nhalf = wv >> 2;

  // Warm this block's W_hh slice into cache (global_prefetch_b8).
  {
    const int c = tid;                                  // 256 cols exactly
    const int g = c >> 6, j = c & 63;
    const int n = g * 320 + blk * 64 + j;
#pragma unroll
    for (int k = 0; k < Hp; k += 64)
      __builtin_prefetch(W + (size_t)n * Hp + k, 0, 1);
  }

  unsigned myphase = 0;
  for (int s = 0; s < T; ++s) {
    const int t = dir ? (T - 1 - s) : s;               // reverse scan for dir 1

    dir_barrier(cnt, phs, NBLK, myphase);              // prev step fully written
    if (wv == 0) tdm_stage_h(hc, h_lds);               // TDM: 40KB -> LDS
    __syncthreads();
    dir_barrier(cnt, phs, NBLK, myphase);              // everyone has read hc

    v8f acc[8] = {};
    const int arow = mtile * 16 + (lane & 15);
    for (int ks = 0; ks < Hp; ks += 32) {
      const v16h a = load_frag_k(h_lds, Hp, arow, ks, lane);   // ds_load path
#pragma unroll
      for (int nt = 0; nt < 8; ++nt) {
        const int c0 = nhalf * 128 + nt * 16;
        const int g = c0 >> 6, j0 = c0 & 63;
        const int n = g * 320 + blk * 64 + j0 + (lane & 15);
        const v16h b = load_frag_k(W, Hp, n, ks, lane);
        acc[nt] = wmma_f16(a, b, acc[nt]);
      }
    }

    const int rlo = (lane >> 4) * 8, col = lane & 15;
    const float* preT = preD + (size_t)t * Bsz * G4;
#pragma unroll
    for (int nt = 0; nt < 8; ++nt) {
      const int c0 = nhalf * 128 + nt * 16;
      const int g = c0 >> 6, j0 = c0 & 63;
      const int n = g * 320 + blk * 64 + j0 + col;
#pragma unroll
      for (int rr = 0; rr < 8; ++rr) {
        const int row = mtile * 16 + rr + rlo;
        gb[row * 256 + c0 + col] = acc[nt][rr] + preT[row * G4 + n];
      }
    }
    __syncthreads();

    for (int e = tid; e < Bsz * 64; e += 256) {
      const int row = e >> 6, j = e & 63;
      const float iv = gb[row * 256 + j];
      const float fv = gb[row * 256 + 64 + j];
      const float gv = gb[row * 256 + 128 + j];
      const float ov = gb[row * 256 + 192 + j];
      const int hj = blk * 64 + j;
      const float cold = cc[row * Hp + hj];
      const float cn = sigmoidf_(fv) * cold + sigmoidf_(iv) * tanhf(gv);
      const float hn = sigmoidf_(ov) * tanhf(cn);
      cc[row * Hp + hj] = cn;
      hc[row * Hp + hj] = (h16)hn;                     // next step's TDM source
      hsD[((size_t)t * Bsz + row) * Hp + hj] = hn;
    }
  }
}

// ----- logits (K=9) + softmax -> probs into d_out --------------------------

__global__ void logits_softmax_kernel(const float* __restrict__ hs,
                                      const float* __restrict__ Wlin,
                                      const float* __restrict__ blin,
                                      float* __restrict__ probs) {
  using namespace cfg;
  const int i = blockIdx.x * blockDim.x + threadIdx.x;   // i = b*T + t
  if (i >= Bsz * T) return;
  const int b = i / T, t = i % T;
  const float* hf = hs + ((size_t)t * Bsz + b) * Hp;
  const float* hb = hs + (size_t)ROWS * Hp + ((size_t)t * Bsz + b) * Hp;
  float lg[Kt];
  float mx = -1e30f;
  for (int k = 0; k < Kt; ++k) {
    const float* wr = Wlin + k * (2 * H);
    float s = blin[k];
    for (int j = 0; j < H; ++j) s += hf[j] * wr[j];
    for (int j = 0; j < H; ++j) s += hb[j] * wr[H + j];
    lg[k] = s;
    mx = fmaxf(mx, s);
  }
  float den = 0.f;
  for (int k = 0; k < Kt; ++k) { lg[k] = __expf(lg[k] - mx); den += lg[k]; }
  const float inv = 1.f / den;
  for (int k = 0; k < Kt; ++k) probs[(size_t)i * Kt + k] = lg[k] * inv;
}

// ----- CRF log-likelihood: one block, one lane per batch row ---------------

__global__ void crf_kernel(const int* __restrict__ y,
                           const float* __restrict__ probs,
                           const float* __restrict__ start,
                           const float* __restrict__ endt,
                           const float* __restrict__ trans,
                           float* __restrict__ out_llh) {
  using namespace cfg;
  __shared__ float tr[Kt * Kt];
  __shared__ float red[Bsz];
  const int b = threadIdx.x;
  for (int i = b; i < Kt * Kt; i += Bsz) tr[i] = trans[i];
  __syncthreads();

  const int* yb = y + b * T;
  const float* em = probs + (size_t)b * T * Kt;

  // numerator (gold path score)
  int prev = yb[0] < 0 ? 0 : yb[0];
  float num = start[prev] + em[prev];
  int seqlen = (yb[0] != -1) ? 1 : 0;
  for (int t = 1; t < T; ++t) {
    const int yt = yb[t];
    const float m = (yt != -1) ? 1.f : 0.f;
    const int tag = yt < 0 ? 0 : yt;
    num += (tr[prev * Kt + tag] + em[t * Kt + tag]) * m;
    seqlen += (yt != -1);
    prev = tag;
  }
  const int li = (seqlen > 0) ? (seqlen - 1) : 0;
  const int last = yb[li] < 0 ? 0 : yb[li];
  num += endt[last];

  // denominator (forward algorithm, 9 states)
  float alpha[Kt];
  for (int k = 0; k < Kt; ++k) alpha[k] = start[k] + em[k];
  for (int t = 1; t < T; ++t) {
    if (yb[t] == -1) continue;
    float na[Kt];
    for (int j = 0; j < Kt; ++j) {
      float mx = -1e30f;
      for (int i = 0; i < Kt; ++i) mx = fmaxf(mx, alpha[i] + tr[i * Kt + j]);
      float s = 0.f;
      for (int i = 0; i < Kt; ++i) s += __expf(alpha[i] + tr[i * Kt + j] - mx);
      na[j] = mx + logf(s) + em[t * Kt + j];
    }
    for (int j = 0; j < Kt; ++j) alpha[j] = na[j];
  }
  float mx = -1e30f;
  for (int j = 0; j < Kt; ++j) mx = fmaxf(mx, alpha[j] + endt[j]);
  float s = 0.f;
  for (int j = 0; j < Kt; ++j) s += __expf(alpha[j] + endt[j] - mx);
  const float den = mx + logf(s);

  red[b] = num - den;
  __syncthreads();
  if (b == 0) {
    float tot = 0.f;
    for (int i = 0; i < Bsz; ++i) tot += red[i];
    out_llh[0] = -tot;
  }
}

// ---------------------------------------------------------------------------

extern "C" void kernel_launch(void* const* d_in, const int* in_sizes, int n_in,
                              void* d_out, int out_size, void* d_ws,
                              size_t ws_size, hipStream_t stream) {
  using namespace cfg;
  (void)in_sizes; (void)n_in; (void)out_size; (void)ws_size;

  const int*   x       = (const int*)  d_in[0];
  const int*   y       = (const int*)  d_in[1];
  const float* embed   = (const float*)d_in[2];
  const float* W_ih_f  = (const float*)d_in[3];
  const float* W_hh_f  = (const float*)d_in[4];
  const float* b_ih_f  = (const float*)d_in[5];
  const float* b_hh_f  = (const float*)d_in[6];
  const float* W_ih_b  = (const float*)d_in[7];
  const float* W_hh_b  = (const float*)d_in[8];
  const float* b_ih_b  = (const float*)d_in[9];
  const float* b_hh_b  = (const float*)d_in[10];
  const float* W_lin   = (const float*)d_in[11];
  const float* b_lin   = (const float*)d_in[12];
  const float* start_t = (const float*)d_in[13];
  const float* end_t   = (const float*)d_in[14];
  const float* trans   = (const float*)d_in[15];

  // ---- workspace carve-up (256B aligned) ----
  char* w = (char*)d_ws;
  auto alloc = [&](size_t bytes) -> void* {
    void* p = (void*)w;
    w += (bytes + 255) & ~(size_t)255;
    return p;
  };
  h16*      embh   = (h16*)     alloc((size_t)ROWS * Ep * sizeof(h16));
  h16*      WihP   = (h16*)     alloc((size_t)2 * G4 * Ep * sizeof(h16));
  h16*      WhhP   = (h16*)     alloc((size_t)2 * G4 * Hp * sizeof(h16));
  float*    bias   = (float*)   alloc((size_t)2 * G4 * sizeof(float));
  float*    pre    = (float*)   alloc((size_t)2 * ROWS * G4 * sizeof(float));
  float*    hsb    = (float*)   alloc((size_t)2 * ROWS * Hp * sizeof(float));
  h16*      h16cur = (h16*)     alloc((size_t)2 * Bsz * Hp * sizeof(h16));
  float*    ccur   = (float*)   alloc((size_t)2 * Bsz * Hp * sizeof(float));
  float*    gbuf   = (float*)   alloc((size_t)2 * NBLK * Bsz * 256 * sizeof(float));
  unsigned* bar    = (unsigned*)alloc(256);

  float* probs = (float*)d_out;                       // (B,T,K)
  float* llh   = probs + (size_t)Bsz * T * Kt;        // final scalar

  init_kernel<<<160, 256, 0, stream>>>(h16cur, ccur, bar);

  prep_weights_kernel<<<2048, 256, 0, stream>>>(
      W_ih_f, W_hh_f, b_ih_f, b_hh_f, W_ih_b, W_hh_b, b_ih_b, b_hh_b,
      WihP, WhhP, bias);

  embed_gather_kernel<<<4096, 256, 0, stream>>>(x, embed, embh);

  // 2 dirs * 2048 M-tiles * 20 N-groups = 81920 waves / 8 waves per block
  gemm_pre_kernel<<<10240, 256, 0, stream>>>(embh, WihP, bias, pre);

  lstm_kernel<<<2 * NBLK, 256, 0, stream>>>(WhhP, pre, hsb, h16cur, ccur, gbuf,
                                            bar);

  logits_softmax_kernel<<<(Bsz * T) / 256, 256, 0, stream>>>(hsb, W_lin, b_lin,
                                                             probs);

  crf_kernel<<<1, Bsz, 0, stream>>>(y, probs, start_t, end_t, trans, llh);
}
